// AdapterBlock3D_65755949302444
// MI455X (gfx1250) — matrix operand order repaired and checked by reading, not compiled, not run on hardware
//
#include <hip/hip_runtime.h>
#include <hip/hip_bf16.h>
#include <math.h>

#define DIMC   768
#define HEADS  12
#define HD     64
#define NWIN   16
#define NTOK   512
#define T_TOT  8192
#define AHD    192
#define MLPD   3072
#define QKV3   2304
#define ATT_SCALE 0.125f

typedef __attribute__((ext_vector_type(16))) __bf16 v16bf;
typedef __attribute__((ext_vector_type(8)))  float  v8f;
typedef __attribute__((ext_vector_type(4)))  unsigned int u32x4;
typedef __attribute__((ext_vector_type(8)))  int i32x8;
typedef __attribute__((ext_vector_type(4)))  int i32x4;
typedef __bf16 bf16;

#if defined(__has_builtin)
#if __has_builtin(__builtin_amdgcn_tensor_load_to_lds) && __has_builtin(__builtin_amdgcn_s_wait_tensorcnt)
#define HAVE_TDM 1
#else
#define HAVE_TDM 0
#endif
#else
#define HAVE_TDM 0
#endif

// ---------------- WMMA helpers ----------------

__device__ __forceinline__ v8f wmma_bf16(v16bf a, v16bf b, v8f c) {
    return __builtin_amdgcn_wmma_f32_16x16x32_bf16(false, a, false, b, (short)0, c, false, false);
}

// A-matrix fragment (16x32 bf16): lane l -> row m=l&15, K chunks {kb..kb+7, kb+16..kb+23}, kb=(l>>4)*8
__device__ __forceinline__ v16bf frag_a(const bf16* base, int row0, int ld, int kofs) {
    int l = threadIdx.x & 31;
    int m = l & 15, kb = (l >> 4) * 8;
    const bf16* p = base + (size_t)(row0 + m) * ld + kofs + kb;
    union { v16bf v; uint4 q[2]; } u;
    u.q[0] = *(const uint4*)p;
    u.q[1] = *(const uint4*)(p + 16);
    return u.v;
}

// B-matrix fragment (32x16 bf16), stored as rows of W (row=n, col=k):
// lane l -> col n=l&15, K chunks {kb..kb+7, kb+16..kb+23}, kb=(l>>4)*8
__device__ __forceinline__ v16bf frag_b(const bf16* base, int row0, int ld, int kofs) {
    int l = threadIdx.x & 31;
    int n = l & 15, k0 = (l >> 4) * 8;
    const bf16* p = base + (size_t)(row0 + n) * ld + kofs + k0;
    union { v16bf v; uint4 q[2]; } u;
    u.q[0] = *(const uint4*)p;
    u.q[1] = *(const uint4*)(p + 16);
    return u.v;
}

// ---------------- Tensor Data Mover: 2D tile -> LDS with row padding ----------------
// D# group0: count=1, lds_addr, 57-bit global addr, type=2 ("image")
// D# group1: data_size=2B, pad_enable, pad codes, tensor dims (OOB rows zero-fill),
//            tile dims, tensor_dim0_stride.

#if HAVE_TDM
__device__ __forceinline__ void tdm_load_2d(unsigned lds_off, const void* gptr,
                                            unsigned tile_rows, unsigned valid_rows,
                                            unsigned row_elems, unsigned stride_elems,
                                            unsigned pad_interval_code, unsigned pad_amount_code) {
    unsigned long long ga = (unsigned long long)(uintptr_t)gptr;
    u32x4 g0 = {0u, 0u, 0u, 0u};
    g0[0] = 1u;                                                  // count=1 (valid descriptor)
    g0[1] = lds_off;                                             // LDS byte address
    g0[2] = (unsigned)ga;                                        // global addr [31:0]
    g0[3] = (unsigned)((ga >> 32) & 0x01ffffffu) | (2u << 30);   // addr [56:32] | type=2
    i32x8 g1 = {0, 0, 0, 0, 0, 0, 0, 0};
    g1[0] = (int)((1u << 16)                                     // data_size: 2 bytes
                | (1u << 20)                                     // pad_enable
                | (pad_interval_code << 22)
                | (pad_amount_code << 25));
    g1[1] = (int)((row_elems & 0xffffu) << 16);                  // tensor_dim0[15:0]
    g1[2] = (int)(((row_elems >> 16) & 0xffffu)                  // tensor_dim0[31:16]
                | ((valid_rows & 0xffffu) << 16));               // tensor_dim1[15:0]
    g1[3] = (int)(((valid_rows >> 16) & 0xffffu)                 // tensor_dim1[31:16]
                | ((row_elems & 0xffffu) << 16));                // tile_dim0
    g1[4] = (int)(tile_rows & 0xffffu);                          // tile_dim1 (tile_dim2=0)
    g1[5] = (int)stride_elems;                                   // tensor_dim0_stride[31:0]
    i32x4 z4 = {0, 0, 0, 0};
#if __clang_major__ >= 23
    i32x8 z8 = {0, 0, 0, 0, 0, 0, 0, 0};
    __builtin_amdgcn_tensor_load_to_lds(g0, g1, z4, z4, z8, 0);
#else
    __builtin_amdgcn_tensor_load_to_lds(g0, g1, z4, z4, 0);
#endif
}
#endif

// ---------------- weight convert ----------------

__global__ void k_f2bf(const float* __restrict__ src, bf16* __restrict__ dst, int n) {
    int i = blockIdx.x * 256 + threadIdx.x;
    if (i < n) dst[i] = (bf16)src[i];
}

// ---------------- LN1 + window partition ----------------

__global__ __launch_bounds__(256)
void k_ln1(const float* __restrict__ x, const float* __restrict__ g,
           const float* __restrict__ b, bf16* __restrict__ out) {
    __shared__ float red[256];
    int t = blockIdx.x, tid = threadIdx.x;
    int bidx = t >> 12, rem = t & 4095;
    int D = rem >> 8, H = (rem >> 4) & 15, Wp = rem & 15;
    int win = ((bidx * 2 + (D >> 3)) * 2 + (H >> 3)) * 2 + (Wp >> 3);
    int n   = ((D & 7) * 8 + (H & 7)) * 8 + (Wp & 7);
    size_t orow = (size_t)(win * NTOK + n) * DIMC;
    const float* xr = x + (size_t)t * DIMC;
    float v0 = xr[tid], v1 = xr[tid + 256], v2 = xr[tid + 512];
    red[tid] = v0 + v1 + v2; __syncthreads();
    for (int o = 128; o > 0; o >>= 1) { if (tid < o) red[tid] += red[tid + o]; __syncthreads(); }
    float mean = red[0] * (1.f / DIMC); __syncthreads();
    float d0 = v0 - mean, d1 = v1 - mean, d2 = v2 - mean;
    red[tid] = d0 * d0 + d1 * d1 + d2 * d2; __syncthreads();
    for (int o = 128; o > 0; o >>= 1) { if (tid < o) red[tid] += red[tid + o]; __syncthreads(); }
    float rstd = rsqrtf(red[0] * (1.f / DIMC) + 1e-5f);
    out[orow + tid]       = (bf16)(d0 * rstd * g[tid]       + b[tid]);
    out[orow + tid + 256] = (bf16)(d1 * rstd * g[tid + 256] + b[tid + 256]);
    out[orow + tid + 512] = (bf16)(d2 * rstd * g[tid + 512] + b[tid + 512]);
}

// ---------------- residual + unpartition + LN2 ----------------

__global__ __launch_bounds__(256)
void k_addln2(const float* __restrict__ x, const float* __restrict__ y2,
              const float* __restrict__ g, const float* __restrict__ b,
              float* __restrict__ h, bf16* __restrict__ hn) {
    __shared__ float red[256];
    int t = blockIdx.x, tid = threadIdx.x;
    int bidx = t >> 12, rem = t & 4095;
    int D = rem >> 8, H = (rem >> 4) & 15, Wp = rem & 15;
    int win = ((bidx * 2 + (D >> 3)) * 2 + (H >> 3)) * 2 + (Wp >> 3);
    int n   = ((D & 7) * 8 + (H & 7)) * 8 + (Wp & 7);
    size_t wrow = (size_t)(win * NTOK + n) * DIMC;
    size_t trow = (size_t)t * DIMC;
    float v0 = x[trow + tid]       + y2[wrow + tid];
    float v1 = x[trow + tid + 256] + y2[wrow + tid + 256];
    float v2 = x[trow + tid + 512] + y2[wrow + tid + 512];
    h[trow + tid] = v0; h[trow + tid + 256] = v1; h[trow + tid + 512] = v2;
    red[tid] = v0 + v1 + v2; __syncthreads();
    for (int o = 128; o > 0; o >>= 1) { if (tid < o) red[tid] += red[tid + o]; __syncthreads(); }
    float mean = red[0] * (1.f / DIMC); __syncthreads();
    float d0 = v0 - mean, d1 = v1 - mean, d2 = v2 - mean;
    red[tid] = d0 * d0 + d1 * d1 + d2 * d2; __syncthreads();
    for (int o = 128; o > 0; o >>= 1) { if (tid < o) red[tid] += red[tid + o]; __syncthreads(); }
    float rstd = rsqrtf(red[0] * (1.f / DIMC) + 1e-5f);
    hn[trow + tid]       = (bf16)(d0 * rstd * g[tid]       + b[tid]);
    hn[trow + tid + 256] = (bf16)(d1 * rstd * g[tid + 256] + b[tid + 256]);
    hn[trow + tid + 512] = (bf16)(d2 * rstd * g[tid + 512] + b[tid + 512]);
}

// ---------------- generic WMMA GEMM: out = resid + alpha*epi(A @ W^T + bias) ----------------

#define BM 128
#define BN 128
#define BKK 64
#define LDT 72   // padded LDS row stride: 144B, 16B-aligned, bank-conflict-free

__global__ __launch_bounds__(256)
void k_gemm(const bf16* __restrict__ A, const bf16* __restrict__ W,
            const float* __restrict__ bias, const float* resid,
            float* outf, bf16* __restrict__ outb,
            int M, int N, int K, float alpha, int do_gelu) {
    __shared__ bf16 As[BM * LDT];
    __shared__ bf16 Bs[BN * LDT];
    int tid = threadIdx.x;
    int lane = tid & 31, wave = tid >> 5;
    int m0 = blockIdx.y * BM;
    int n0 = blockIdx.x * BN;
    int wm = (wave >> 2) * 64;
    int wn = (wave & 3) * 32;

    v8f zero = {};
    v8f acc[4][2];
    #pragma unroll
    for (int i = 0; i < 4; i++)
        for (int j = 0; j < 2; j++) acc[i][j] = zero;

    int bn_valid = N - n0; if (bn_valid > BN) bn_valid = BN;
#if HAVE_TDM
    unsigned as_ofs = (unsigned)(uintptr_t)(void*)As;
    unsigned bs_ofs = (unsigned)(uintptr_t)(void*)Bs;
#endif

    for (int k0 = 0; k0 < K; k0 += BKK) {
#if HAVE_TDM
        // 64-elem rows (128B = 32 DW) -> pad_interval code 4; pad 4 DW -> code 3 (stride 72 elems)
        if (wave == 0) {
            tdm_load_2d(as_ofs, A + (size_t)m0 * K + k0, BM, BM, BKK, (unsigned)K, 4, 3);
            tdm_load_2d(bs_ofs, W + (size_t)n0 * K + k0, BN, (unsigned)bn_valid, BKK, (unsigned)K, 4, 3);
            __builtin_amdgcn_s_wait_tensorcnt(0);
        }
#else
        int lrow = tid >> 2, lseg = (tid & 3) * 16;
        const bf16* ga = A + (size_t)(m0 + lrow) * K + k0 + lseg;
        uint4 a0 = *(const uint4*)ga;
        uint4 a1 = *(const uint4*)(ga + 8);
        *(uint4*)&As[lrow * LDT + lseg]     = a0;
        *(uint4*)&As[lrow * LDT + lseg + 8] = a1;
        uint4 b0 = {}, b1 = {};
        if (lrow < bn_valid) {
            const bf16* gb = W + (size_t)(n0 + lrow) * K + k0 + lseg;
            b0 = *(const uint4*)gb;
            b1 = *(const uint4*)(gb + 8);
        }
        *(uint4*)&Bs[lrow * LDT + lseg]     = b0;
        *(uint4*)&Bs[lrow * LDT + lseg + 8] = b1;
#endif
        if (k0 + BKK < K)
            __builtin_prefetch(A + (size_t)(m0 + (tid & 127)) * K + k0 + BKK, 0, 3);
        __syncthreads();

        #pragma unroll
        for (int kk = 0; kk < BKK; kk += 32) {
            v16bf bfrag[2];
            #pragma unroll
            for (int j = 0; j < 2; j++) bfrag[j] = frag_b(Bs, wn + j * 16, LDT, kk);
            #pragma unroll
            for (int i = 0; i < 4; i++) {
                v16bf afrag = frag_a(As, wm + i * 16, LDT, kk);
                #pragma unroll
                for (int j = 0; j < 2; j++)
                    acc[i][j] = wmma_bf16(afrag, bfrag[j], acc[i][j]);
            }
        }
        __syncthreads();
    }

    int hlf = lane >> 4, nl = lane & 15;
    #pragma unroll
    for (int i = 0; i < 4; i++) {
        #pragma unroll
        for (int j = 0; j < 2; j++) {
            int col = n0 + wn + j * 16 + nl;
            if (col >= N) continue;
            float bb = bias ? bias[col] : 0.f;
            #pragma unroll
            for (int r = 0; r < 8; r++) {
                int row = m0 + wm + i * 16 + hlf * 8 + r;
                float v = acc[i][j][r] + bb;
                if (do_gelu) v = 0.5f * v * (1.f + erff(v * 0.70710678118f));
                v *= alpha;
                if (resid) v += resid[(size_t)row * N + col];
                if (outf) outf[(size_t)row * N + col] = v;
                if (outb) outb[(size_t)row * N + col] = (bf16)v;
            }
        }
    }
}

// ---------------- attention (flash-style, per window-head) ----------------

#define ATT_LDK 72   // 144B rows, matches TDM pad codes (4,3)
#define ATT_LDV 56
#define ATT_LDP 56

__global__ __launch_bounds__(256)
void k_attn(const bf16* __restrict__ qkv, const float* __restrict__ rpd,
            const float* __restrict__ rph, const float* __restrict__ rpw,
            bf16* __restrict__ out) {
    __shared__ bf16  Kt[32 * ATT_LDK];
    __shared__ bf16  Vt[HD * ATT_LDV];       // V transposed: [hd][key]
    __shared__ bf16  Pb[8][16 * ATT_LDP];    // per-wave P staging
    __shared__ float relb[8][16][24];        // per-wave rel-pos bias (d:0-7,h:8-15,w:16-23)

    int bx = blockIdx.x;
    int qt = bx & 3;
    int wh = bx >> 2;
    int head = wh % HEADS;
    int win  = wh / HEADS;
    int tid = threadIdx.x, lane = tid & 31, wave = tid >> 5;
    int q0 = qt * 128 + wave * 16;

    const bf16* qbase = qkv + (size_t)(win * NTOK) * QKV3 + head * HD;
    const bf16* kbase = qbase + DIMC;
    const bf16* vbase = qbase + 2 * DIMC;

    // decomposed rel-pos bias: relb[m][axis*8+j] = dot(q_m, table[pos-j+7])
    for (int task = lane; task < 16 * 24; task += 32) {
        int m = task / 24, aj = task % 24, axis = aj >> 3, j = aj & 7;
        int qrow = q0 + m;
        int pos = (axis == 0) ? (qrow >> 6) : (axis == 1) ? ((qrow >> 3) & 7) : (qrow & 7);
        const float* tab = (axis == 0 ? rpd : axis == 1 ? rph : rpw) + (size_t)(pos - j + 7) * HD;
        const bf16* qr = qbase + (size_t)qrow * QKV3;
        float s = 0.f;
        for (int c = 0; c < HD; c++) s += (float)qr[c] * tab[c];
        relb[wave][m][aj] = s;
    }

    v16bf qfrag[2];
    qfrag[0] = frag_a(qbase, q0, QKV3, 0);
    qfrag[1] = frag_a(qbase, q0, QKV3, 32);

    v8f zero = {};
    v8f oacc[4];
    #pragma unroll
    for (int i = 0; i < 4; i++) oacc[i] = zero;
    float mrun[8], lrun[8];
    #pragma unroll
    for (int r = 0; r < 8; r++) { mrun[r] = -1e30f; lrun[r] = 0.f; }

    int hlf = lane >> 4, nl = lane & 15;
#if HAVE_TDM
    unsigned kt_ofs = (unsigned)(uintptr_t)(void*)Kt;
#endif

    for (int kt = 0; kt < NTOK; kt += 32) {
        __syncthreads();
#if HAVE_TDM
        if (wave == 0)
            tdm_load_2d(kt_ofs, kbase + (size_t)kt * QKV3, 32, 32, HD, QKV3, 4, 3);
#endif
        {   // V tile transposed (all threads); K tile manual only without TDM
            int row = tid >> 3, seg = (tid & 7) * 8;
#if !HAVE_TDM
            uint4 dk = *(const uint4*)(kbase + (size_t)(kt + row) * QKV3 + seg);
            *(uint4*)&Kt[row * ATT_LDK + seg] = dk;
#endif
            uint4 dv = *(const uint4*)(vbase + (size_t)(kt + row) * QKV3 + seg);
            const bf16* pv = (const bf16*)&dv;
            #pragma unroll
            for (int e = 0; e < 8; e++) Vt[(seg + e) * ATT_LDV + row] = pv[e];
        }
#if HAVE_TDM
        if (wave == 0) __builtin_amdgcn_s_wait_tensorcnt(0);
#endif
        __syncthreads();

        // S = scale * Q K^T + bias  (two 16-col tiles)
        v8f s[2];
        #pragma unroll
        for (int j = 0; j < 2; j++) {
            v8f sc = zero;
            sc = wmma_bf16(qfrag[0], frag_b(Kt, j * 16, ATT_LDK, 0), sc);
            sc = wmma_bf16(qfrag[1], frag_b(Kt, j * 16, ATT_LDK, 32), sc);
            s[j] = sc;
        }

        float tmax[8];
        #pragma unroll
        for (int r = 0; r < 8; r++) {
            int mm = hlf * 8 + r;
            int key = kt + nl;
            float bs0 = relb[wave][mm][key >> 6] + relb[wave][mm][8 + ((key >> 3) & 7)]
                      + relb[wave][mm][16 + (key & 7)];
            key += 16;
            float bs1 = relb[wave][mm][key >> 6] + relb[wave][mm][8 + ((key >> 3) & 7)]
                      + relb[wave][mm][16 + (key & 7)];
            s[0][r] = s[0][r] * ATT_SCALE + bs0;
            s[1][r] = s[1][r] * ATT_SCALE + bs1;
            float t = fmaxf(s[0][r], s[1][r]);
            for (int msk = 1; msk < 16; msk <<= 1) t = fmaxf(t, __shfl_xor(t, msk, 32));
            tmax[r] = t;
        }

        // online softmax
        #pragma unroll
        for (int r = 0; r < 8; r++) {
            float mnew = fmaxf(mrun[r], tmax[r]);
            float corr = __expf(mrun[r] - mnew);
            mrun[r] = mnew;
            lrun[r] *= corr;
            #pragma unroll
            for (int i = 0; i < 4; i++) oacc[i][r] *= corr;
            float p0 = __expf(s[0][r] - mnew);
            float p1 = __expf(s[1][r] - mnew);
            s[0][r] = p0; s[1][r] = p1;
            float ps = p0 + p1;
            for (int msk = 1; msk < 16; msk <<= 1) ps += __shfl_xor(ps, msk, 32);
            lrun[r] += ps;
        }

        // P -> LDS (C layout -> A layout relayout), then O += P @ V
        #pragma unroll
        for (int r = 0; r < 8; r++) {
            int mm = hlf * 8 + r;
            Pb[wave][mm * ATT_LDP + nl]      = (bf16)s[0][r];
            Pb[wave][mm * ATT_LDP + 16 + nl] = (bf16)s[1][r];
        }
        v16bf pfrag = frag_a(Pb[wave], 0, ATT_LDP, 0);
        #pragma unroll
        for (int i = 0; i < 4; i++)
            oacc[i] = wmma_bf16(pfrag, frag_b(Vt, i * 16, ATT_LDV, 0), oacc[i]);
    }

    #pragma unroll
    for (int i = 0; i < 4; i++) {
        #pragma unroll
        for (int r = 0; r < 8; r++) {
            int row = q0 + hlf * 8 + r;
            int hd = i * 16 + nl;
            float v = oacc[i][r] / lrun[r];
            out[(size_t)(win * NTOK + row) * DIMC + head * HD + hd] = (bf16)v;
        }
    }
}

// ---------------- host launcher ----------------

extern "C" void kernel_launch(void* const* d_in, const int* in_sizes, int n_in,
                              void* d_out, int out_size, void* d_ws, size_t ws_size,
                              hipStream_t stream) {
    (void)in_sizes; (void)n_in; (void)out_size; (void)ws_size;
    char* ws = (char*)d_ws;
    size_t off = 0;
    auto take = [&](size_t bytes) -> void* {
        void* p = ws + off;
        off += (bytes + 255) & ~(size_t)255;
        return p;
    };

    const size_t T = T_TOT;
    bf16* wqkv  = (bf16*)take((size_t)QKV3 * DIMC * 2);
    bf16* wproj = (bf16*)take((size_t)DIMC * DIMC * 2);
    bf16* waa1  = (bf16*)take((size_t)AHD  * DIMC * 2);
    bf16* waa2  = (bf16*)take((size_t)DIMC * AHD  * 2);
    bf16* wmlp1 = (bf16*)take((size_t)MLPD * DIMC * 2);
    bf16* wmlp2 = (bf16*)take((size_t)DIMC * MLPD * 2);
    bf16* wma1  = (bf16*)take((size_t)AHD  * DIMC * 2);
    bf16* wma2  = (bf16*)take((size_t)DIMC * AHD  * 2);

    // region1: [ln1out | qkvb] = T*3072 bf16; later reused as y1b (front) and m1 (whole)
    bf16*  region1 = (bf16*)take(T * 3072 * 2);
    bf16*  ln1out  = region1;
    bf16*  qkvb    = region1 + T * DIMC;
    bf16*  y1b     = region1;
    bf16*  m1      = region1;
    bf16*  attnout = (bf16*)take(T * DIMC * 2);
    bf16*  hn      = attnout;            // reuse: attnout dead after proj
    float* y1      = (float*)take(T * DIMC * 4);
    bf16*  a1      = (bf16*)take(T * AHD * 2);
    float* y2      = (float*)take(T * DIMC * 4);
    float* hbuf    = (float*)take(T * DIMC * 4);
    bf16*  am1     = (bf16*)take(T * AHD * 2);

    auto conv = [&](const void* src, bf16* dst, int n) {
        k_f2bf<<<(n + 255) / 256, 256, 0, stream>>>((const float*)src, dst, n);
    };
    conv(d_in[3],  wqkv,  QKV3 * DIMC);
    conv(d_in[8],  wproj, DIMC * DIMC);
    conv(d_in[10], waa1,  AHD * DIMC);
    conv(d_in[12], waa2,  DIMC * AHD);
    conv(d_in[16], wmlp1, MLPD * DIMC);
    conv(d_in[18], wmlp2, DIMC * MLPD);
    conv(d_in[20], wma1,  AHD * DIMC);
    conv(d_in[22], wma2,  DIMC * AHD);

    auto gemm = [&](const bf16* A, const bf16* W, const float* bias, const float* resid,
                    float* outf, bf16* outb, int M, int N, int K, float alpha, int gelu) {
        dim3 grid((N + BN - 1) / BN, (M + BM - 1) / BM);
        k_gemm<<<grid, 256, 0, stream>>>(A, W, bias, resid, outf, outb, M, N, K, alpha, gelu);
    };

    // 2. LN1 + window partition
    k_ln1<<<T_TOT, 256, 0, stream>>>((const float*)d_in[0], (const float*)d_in[1],
                                     (const float*)d_in[2], ln1out);
    // 3. QKV projection
    gemm(ln1out, wqkv, (const float*)d_in[4], nullptr, nullptr, qkvb,
         T_TOT, QKV3, DIMC, 1.f, 0);
    // 4. windowed attention with decomposed rel-pos
    k_attn<<<NWIN * HEADS * 4, 256, 0, stream>>>(qkvb, (const float*)d_in[5],
                                                 (const float*)d_in[6],
                                                 (const float*)d_in[7], attnout);
    // 5. output projection
    gemm(attnout, wproj, (const float*)d_in[9], nullptr, y1, y1b,
         T_TOT, DIMC, DIMC, 1.f, 0);
    // 6-7. attention adapter (skip=True)
    gemm(y1b, waa1, (const float*)d_in[11], nullptr, nullptr, a1,
         T_TOT, AHD, DIMC, 1.f, 1);
    gemm(a1, waa2, (const float*)d_in[13], y1, y2, nullptr,
         T_TOT, DIMC, AHD, 1.f, 0);
    // 8. unpartition + shortcut + LN2
    k_addln2<<<T_TOT, 256, 0, stream>>>((const float*)d_in[0], y2,
                                        (const float*)d_in[14], (const float*)d_in[15],
                                        hbuf, hn);
    // 9. MLP fc1 (GELU)
    gemm(hn, wmlp1, (const float*)d_in[17], nullptr, nullptr, m1,
         T_TOT, MLPD, DIMC, 1.f, 1);
    // 10. MLP adapter fc1 (GELU)
    gemm(hn, wma1, (const float*)d_in[21], nullptr, nullptr, am1,
         T_TOT, AHD, DIMC, 1.f, 1);
    // 11. MLP fc2 + residual h  -> d_out
    gemm(m1, wmlp2, (const float*)d_in[19], hbuf, (float*)d_out, nullptr,
         T_TOT, DIMC, MLPD, 1.f, 0);
    // 12. d_out += 0.5 * MLP-adapter fc2
    gemm(am1, wma2, (const float*)d_in[23], (const float*)d_out, (float*)d_out, nullptr,
         T_TOT, DIMC, AHD, 0.5f, 0);
}
